// AxialSpaceTimeTransformer_14972255994142
// MI455X (gfx1250) — compile-verified, matrix operand order checked by reading, not run on hardware
//
#include <hip/hip_runtime.h>
#include <hip/hip_bf16.h>

// ---------------- model constants ----------------
#define DIMV   768
#define DEPTH  8
#define HEADS  12
#define DHV    64
#define DFFV   2048
#define BV     2
#define TV     32
#define SV     256
#define NTOK   (BV*TV*SV)          // 16384 tokens
#define SOFTCL 50.0f

// ---------------- WMMA types ----------------
typedef __attribute__((ext_vector_type(16))) __bf16 v16bf;
typedef __attribute__((ext_vector_type(8)))  __bf16 v8bf;
typedef __attribute__((ext_vector_type(8)))  float  v8f;

__device__ __forceinline__ __bf16 to_bf16(float f) {
  unsigned u = __builtin_bit_cast(unsigned, f);
  unsigned r = u + 0x7FFFu + ((u >> 16) & 1u);   // round-to-nearest-even
  unsigned short h = (unsigned short)(r >> 16);
  return __builtin_bit_cast(__bf16, h);
}
__device__ __forceinline__ float bf16_to_f(__bf16 b) {
  unsigned u = ((unsigned)__builtin_bit_cast(unsigned short, b)) << 16;
  return __builtin_bit_cast(float, u);
}
__device__ __forceinline__ v8f vzero() {
  v8f v;
#pragma unroll
  for (int i = 0; i < 8; i++) v[i] = 0.f;
  return v;
}
__device__ __forceinline__ v8f wmma_bf16(v16bf a, v16bf b, v8f c) {
  return __builtin_amdgcn_wmma_f32_16x16x32_bf16(false, a, false, b, (short)0, c,
                                                 false, false);
}

// ---------------- GEMM tiling ----------------
#define GBM 128
#define GBN 64
#define GBK 64    // two 32-wide WMMA K-slabs per barrier round
#define LSTR 72   // padded LDS row stride (bf16); 144B rows keep 16B-aligned chunks

// ISA 16-bit A/B fragment for K-slab ks (0/1): lanes 0-15 row m, halves =
// K 0..7 | 16..23 ; lanes 16-31 row m, halves = K 8..15 | 24..31 (slab-relative).
// Tiles stored [row][k] padded, rows 64 wide.
__device__ __forceinline__ v16bf load_frag(const __bf16* tile, int rowbase, int lane,
                                           int ks) {
  int r  = rowbase + (lane & 15);
  int kb = ((lane & 16) ? 8 : 0) + ks * 32;
  const __bf16* p = tile + r * LSTR + kb;
  v8bf lo = *reinterpret_cast<const v8bf*>(p);
  v8bf hi = *reinterpret_cast<const v8bf*>(p + 16);
  v16bf f;
#pragma unroll
  for (int i = 0; i < 8; i++) { f[i] = lo[i]; f[i + 8] = hi[i]; }
  return f;
}

// Stage A tile (bf16 source, row-major MxK): 128x64 -> LDS, 16B chunks.
// Also prefetches the next K-slab of A (global_prefetch path).
__device__ __forceinline__ void stage_A(__bf16* Al, const __bf16* A, long m0, int k0,
                                        int K, int tid) {
  const bool pf = (k0 + GBK) < K;
#pragma unroll
  for (int it = 0; it < 4; it++) {
    int idx = tid + it * 256;          // 1024 chunks of 8 bf16
    int row = idx >> 3;
    int kc  = (idx & 7) << 3;
    const __bf16* src = A + (m0 + row) * (long)K + k0 + kc;
    *(uint4*)&Al[row * LSTR + kc] = *(const uint4*)src;   // 16 bytes
    if (pf) __builtin_prefetch(src + GBK, 0, 1);
  }
}

// Stage B tile (f32 source, row-major KxN) transposed into LDS [n][k], cvt to bf16
__device__ __forceinline__ void stage_B(__bf16* Bl, const float* B, int k0, long n0,
                                        int ldb, int tid) {
#pragma unroll
  for (int it = 0; it < 4; it++) {
    int idx = tid + it * 256;          // 1024 vec4 chunks (64x64 f32)
    int kr = idx >> 4;
    int nc = (idx & 15) << 2;
    float4 v = *(const float4*)(B + (long)(k0 + kr) * ldb + n0 + nc);
    Bl[(nc + 0) * LSTR + kr] = to_bf16(v.x);
    Bl[(nc + 1) * LSTR + kr] = to_bf16(v.y);
    Bl[(nc + 2) * LSTR + kr] = to_bf16(v.z);
    Bl[(nc + 3) * LSTR + kr] = to_bf16(v.w);
  }
}

// C[M,N] (+= / =) A_bf16[M,K] x B_f32[K,N] (+bias). 256 thr = 8 waves, 4x2 wave grid.
template <bool ACCUM>
__global__ __launch_bounds__(256, 1)
void gemm_kernel(const __bf16* __restrict__ A, const float* __restrict__ B,
                 const float* __restrict__ bias, float* __restrict__ C,
                 int M, int N, int K) {
  __shared__ __bf16 Al[GBM * LSTR];
  __shared__ __bf16 Bl[GBN * LSTR];
  int tid = threadIdx.x, lane = tid & 31, wid = tid >> 5;
  int wy = wid & 3, wx = wid >> 2;
  long m0 = (long)blockIdx.y * GBM;
  long n0 = (long)blockIdx.x * GBN;
  v8f acc[2][2];
#pragma unroll
  for (int i = 0; i < 2; i++)
#pragma unroll
    for (int j = 0; j < 2; j++) acc[i][j] = vzero();

  for (int k0 = 0; k0 < K; k0 += GBK) {
    stage_A(Al, A, m0, k0, K, tid);
    stage_B(Bl, B, k0, n0, N, tid);
    __syncthreads();
#pragma unroll
    for (int ks = 0; ks < 2; ks++) {
      v16bf a0 = load_frag(Al, wy * 32, lane, ks);
      v16bf a1 = load_frag(Al, wy * 32 + 16, lane, ks);
      v16bf b0 = load_frag(Bl, wx * 32, lane, ks);
      v16bf b1 = load_frag(Bl, wx * 32 + 16, lane, ks);
      acc[0][0] = wmma_bf16(a0, b0, acc[0][0]);
      acc[0][1] = wmma_bf16(a0, b1, acc[0][1]);
      acc[1][0] = wmma_bf16(a1, b0, acc[1][0]);
      acc[1][1] = wmma_bf16(a1, b1, acc[1][1]);
    }
    __syncthreads();
  }
  // C/D layout: reg r, lane L -> m = r + (L>=16?8:0), n = L&15
  int nl = lane & 15;
  int mb = (lane & 16) ? 8 : 0;
#pragma unroll
  for (int ti = 0; ti < 2; ti++)
#pragma unroll
    for (int tj = 0; tj < 2; tj++) {
      v8f a = acc[ti][tj];
#pragma unroll
      for (int r = 0; r < 8; r++) {
        long gm = m0 + wy * 32 + ti * 16 + mb + r;
        long gn = n0 + wx * 32 + tj * 16 + nl;
        float val = a[r];
        if (bias) val += bias[gn];
        if (ACCUM) C[gm * N + gn] += val;
        else       C[gm * N + gn] = val;
      }
    }
}

// FFN-1 fused GLU: computes a = tn@Win[:, j] + b, g = tn@Win[:, DFF+j] + b,
// writes act = a * gelu_exact(g) as bf16 (M x DFF).
__global__ __launch_bounds__(256, 1)
void gemm_ff1_kernel(const __bf16* __restrict__ A, const float* __restrict__ Win,
                     const float* __restrict__ b_in, __bf16* __restrict__ act) {
  __shared__ __bf16 Al[GBM * LSTR];
  __shared__ __bf16 Ba[GBN * LSTR];
  __shared__ __bf16 Bg[GBN * LSTR];
  const int K = DIMV, LDB = 2 * DFFV;
  int tid = threadIdx.x, lane = tid & 31, wid = tid >> 5;
  int wy = wid & 3, wx = wid >> 2;
  long m0 = (long)blockIdx.y * GBM;
  long n0 = (long)blockIdx.x * GBN;
  v8f aA[2][2], aG[2][2];
#pragma unroll
  for (int i = 0; i < 2; i++)
#pragma unroll
    for (int j = 0; j < 2; j++) { aA[i][j] = vzero(); aG[i][j] = vzero(); }

  for (int k0 = 0; k0 < K; k0 += GBK) {
    stage_A(Al, A, m0, k0, K, tid);
    stage_B(Ba, Win, k0, n0, LDB, tid);
    stage_B(Bg, Win, k0, n0 + DFFV, LDB, tid);
    __syncthreads();
#pragma unroll
    for (int ks = 0; ks < 2; ks++) {
      v16bf a0 = load_frag(Al, wy * 32, lane, ks);
      v16bf a1 = load_frag(Al, wy * 32 + 16, lane, ks);
      v16bf ba0 = load_frag(Ba, wx * 32, lane, ks);
      v16bf ba1 = load_frag(Ba, wx * 32 + 16, lane, ks);
      v16bf bg0 = load_frag(Bg, wx * 32, lane, ks);
      v16bf bg1 = load_frag(Bg, wx * 32 + 16, lane, ks);
      aA[0][0] = wmma_bf16(a0, ba0, aA[0][0]);
      aA[0][1] = wmma_bf16(a0, ba1, aA[0][1]);
      aA[1][0] = wmma_bf16(a1, ba0, aA[1][0]);
      aA[1][1] = wmma_bf16(a1, ba1, aA[1][1]);
      aG[0][0] = wmma_bf16(a0, bg0, aG[0][0]);
      aG[0][1] = wmma_bf16(a0, bg1, aG[0][1]);
      aG[1][0] = wmma_bf16(a1, bg0, aG[1][0]);
      aG[1][1] = wmma_bf16(a1, bg1, aG[1][1]);
    }
    __syncthreads();
  }
  int nl = lane & 15;
  int mb = (lane & 16) ? 8 : 0;
#pragma unroll
  for (int ti = 0; ti < 2; ti++)
#pragma unroll
    for (int tj = 0; tj < 2; tj++) {
      v8f va = aA[ti][tj], vg = aG[ti][tj];
#pragma unroll
      for (int r = 0; r < 8; r++) {
        long gm = m0 + wy * 32 + ti * 16 + mb + r;
        long gn = n0 + wx * 32 + tj * 16 + nl;
        float av = va[r] + b_in[gn];
        float gv = vg[r] + b_in[DFFV + gn];
        float ge = 0.5f * gv * (1.0f + erff(gv * 0.70710678118f));  // exact gelu
        act[gm * DFFV + gn] = to_bf16(av * ge);
      }
    }
}

// ---------------- normalization kernels ----------------
__global__ __launch_bounds__(256, 1)
void rmsnorm_bf16_kernel(const float* __restrict__ x, const float* __restrict__ w,
                         __bf16* __restrict__ out) {
  __shared__ float red[256];
  long tok = blockIdx.x;
  const float* xp = x + tok * DIMV;
  int tid = threadIdx.x;
  float v0 = xp[tid], v1 = xp[tid + 256], v2 = xp[tid + 512];
  red[tid] = v0 * v0 + v1 * v1 + v2 * v2;
  __syncthreads();
  for (int s = 128; s > 0; s >>= 1) { if (tid < s) red[tid] += red[tid + s]; __syncthreads(); }
  float inv = rsqrtf(red[0] * (1.0f / DIMV) + 1e-6f);
  __bf16* op = out + tok * DIMV;
  op[tid]       = to_bf16(v0 * inv * w[tid]);
  op[tid + 256] = to_bf16(v1 * inv * w[tid + 256]);
  op[tid + 512] = to_bf16(v2 * inv * w[tid + 512]);
}

__global__ __launch_bounds__(256, 1)
void rmsnorm_f32_kernel(const float* __restrict__ x, const float* __restrict__ w,
                        float* __restrict__ out) {
  __shared__ float red[256];
  long tok = blockIdx.x;
  const float* xp = x + tok * DIMV;
  int tid = threadIdx.x;
  float v0 = xp[tid], v1 = xp[tid + 256], v2 = xp[tid + 512];
  red[tid] = v0 * v0 + v1 * v1 + v2 * v2;
  __syncthreads();
  for (int s = 128; s > 0; s >>= 1) { if (tid < s) red[tid] += red[tid + s]; __syncthreads(); }
  float inv = rsqrtf(red[0] * (1.0f / DIMV) + 1e-6f);
  float* op = out + tok * DIMV;
  op[tid]       = v0 * inv * w[tid];
  op[tid + 256] = v1 * inv * w[tid + 256];
  op[tid + 512] = v2 * inv * w[tid + 512];
}

// ---------------- per-(token,head) small kernels ----------------
// mix = sigmoid(tn . Wmix[:,h] + bm[h]); gate = sigmoid(tn . Wg[:,h])
__global__ __launch_bounds__(32, 1)
void mixgate_kernel(const __bf16* __restrict__ tn, const float* __restrict__ Wmix,
                    const float* __restrict__ bmix, const float* __restrict__ Wg,
                    float* __restrict__ mix, float* __restrict__ gate) {
  int lane = threadIdx.x;
  int h = blockIdx.x % HEADS;
  long tok = blockIdx.x / HEADS;
  const __bf16* tp = tn + tok * DIMV;
  float sm = 0.f, sg = 0.f;
#pragma unroll
  for (int r = 0; r < DIMV / 32; r++) {
    int d = lane + r * 32;
    float tv = bf16_to_f(tp[d]);
    sm += tv * Wmix[d * HEADS + h];
    sg += tv * Wg[d * HEADS + h];
  }
  for (int m = 16; m > 0; m >>= 1) { sm += __shfl_xor(sm, m, 32); sg += __shfl_xor(sg, m, 32); }
  if (lane == 0) {
    mix[tok * HEADS + h]  = 1.f / (1.f + __expf(-(sm + bmix[h])));
    gate[tok * HEADS + h] = 1.f / (1.f + __expf(-sg));
  }
}

// v <- v + mix * (rv - v)   (value-residual lerp)
__global__ __launch_bounds__(256, 1)
void vmix_kernel(float* __restrict__ v, const float* __restrict__ rv,
                 const float* __restrict__ mix) {
  long idx = (long)blockIdx.x * 256 + threadIdx.x;     // over NTOK*768
  long tok = idx / DIMV;
  int h = (int)((idx % DIMV) >> 6);
  float m = mix[tok * HEADS + h];
  float vv = v[idx];
  v[idx] = vv + m * (rv[idx] - vv);
}

// k <- l2norm(k_head) * (gamma+1)*sqrt(DH); rotary on q,k for time layers.
// one wave per (token, head); lane owns dims {lane, lane+32} (rotary half-pair)
__global__ __launch_bounds__(32, 1)
void qkprep_kernel(float* __restrict__ q, float* __restrict__ k,
                   const float* __restrict__ kgam, int time_layer) {
  int lane = threadIdx.x;
  int h = blockIdx.x % HEADS;
  long tok = blockIdx.x / HEADS;
  float* kp = k + tok * DIMV + h * DHV;
  float k1 = kp[lane], k2 = kp[lane + 32];
  float ss = k1 * k1 + k2 * k2;
  for (int m = 16; m > 0; m >>= 1) ss += __shfl_xor(ss, m, 32);
  float inv = 1.f / fmaxf(sqrtf(ss), 1e-12f);
  float g1 = (kgam[h * DHV + lane] + 1.f) * 8.f;        // sqrt(64)=8
  float g2 = (kgam[h * DHV + lane + 32] + 1.f) * 8.f;
  k1 = k1 * inv * g1;
  k2 = k2 * inv * g2;
  if (time_layer) {
    int t = (int)((tok / SV) % TV);
    float invf = __expf(-(float)lane * (9.21034037198f / 32.0f));  // 10000^(-lane/32)
    float ang = (float)t * invf;
    float sn, cs;
    __sincosf(ang, &sn, &cs);
    float nk1 = k1 * cs - k2 * sn, nk2 = k2 * cs + k1 * sn;
    k1 = nk1; k2 = nk2;
    float* qp = q + tok * DIMV + h * DHV;
    float q1 = qp[lane], q2 = qp[lane + 32];
    qp[lane]      = q1 * cs - q2 * sn;
    qp[lane + 32] = q2 * cs + q1 * sn;
  }
  kp[lane] = k1;
  kp[lane + 32] = k2;
}

// ---------------- attention core ----------------
// One block per (sequence, head); blockDim.x == n. K/V in dynamic LDS.
// Online softmax with tanh soft-clamp; gate applied on write; bf16 output.
__global__ __launch_bounds__(256, 1)
void attn_kernel(const float* __restrict__ q, const float* __restrict__ k,
                 const float* __restrict__ v, const float* __restrict__ gate,
                 __bf16* __restrict__ o, int n, int stride_tok, int causal,
                 int time_mode) {
  extern __shared__ float smem[];
  float* Ksh = smem;
  float* Vsh = smem + (long)n * DHV;
  int h = blockIdx.x % HEADS;
  int sid = blockIdx.x / HEADS;
  long base;
  if (time_mode) { int b = sid / SV; int s = sid % SV; base = (long)b * TV * SV + s; }
  else           { base = (long)sid * SV; }
  int i = threadIdx.x;
  long tok_i = base + (long)i * stride_tok;
  // cooperative K/V load: thread i loads row i
  {
    const float4* kp = (const float4*)(k + tok_i * DIMV + h * DHV);
    const float4* vp = (const float4*)(v + tok_i * DIMV + h * DHV);
    float4* kd = (float4*)(Ksh + i * DHV);
    float4* vd = (float4*)(Vsh + i * DHV);
#pragma unroll
    for (int c = 0; c < DHV / 4; c++) { kd[c] = kp[c]; vd[c] = vp[c]; }
  }
  __syncthreads();
  float qr[DHV];
  {
    const float* qp = q + tok_i * DIMV + h * DHV;
#pragma unroll
    for (int d = 0; d < DHV; d++) qr[d] = qp[d] * 0.125f;   // DH^-0.5
  }
  float m = -3.0e38f, l = 0.f;
  float oa[DHV];
#pragma unroll
  for (int d = 0; d < DHV; d++) oa[d] = 0.f;
  int jmax = causal ? (i + 1) : n;
  for (int j = 0; j < jmax; j++) {
    float s = 0.f;
    const float* kr = Ksh + j * DHV;
#pragma unroll
    for (int d = 0; d < DHV; d++) s += qr[d] * kr[d];
    s = tanhf(s * (1.0f / SOFTCL)) * SOFTCL;
    float nm = fmaxf(m, s);
    float sc = __expf(m - nm);
    float p  = __expf(s - nm);
    l = l * sc + p;
    const float* vr = Vsh + j * DHV;
#pragma unroll
    for (int d = 0; d < DHV; d++) oa[d] = oa[d] * sc + p * vr[d];
    m = nm;
  }
  float g = gate[tok_i * HEADS + h] / l;
  __bf16* op = o + tok_i * DIMV + h * DHV;
#pragma unroll
  for (int d = 0; d < DHV; d++) op[d] = to_bf16(oa[d] * g);
}

// ---------------- host orchestration ----------------
extern "C" void kernel_launch(void* const* d_in, const int* in_sizes, int n_in,
                              void* d_out, int out_size, void* d_ws, size_t ws_size,
                              hipStream_t stream) {
  (void)in_sizes; (void)n_in; (void)out_size; (void)ws_size;
  const float* tokens      = (const float*)d_in[0];
  const float* attn_norm_w = (const float*)d_in[1];
  const float* Wq          = (const float*)d_in[2];
  const float* Wk          = (const float*)d_in[3];
  const float* Wv          = (const float*)d_in[4];
  const float* Wo          = (const float*)d_in[5];
  const float* Wg          = (const float*)d_in[6];
  const float* Wmix        = (const float*)d_in[7];
  const float* bmix        = (const float*)d_in[8];
  const float* kgam        = (const float*)d_in[9];
  const float* ff_norm_w   = (const float*)d_in[10];
  const float* Win         = (const float*)d_in[11];
  const float* b_in        = (const float*)d_in[12];
  const float* Wout        = (const float*)d_in[13];
  const float* b_out       = (const float*)d_in[14];
  const float* vr_norm_w   = (const float*)d_in[15];
  const float* vr_W        = (const float*)d_in[16];
  const float* final_norm_w= (const float*)d_in[17];

  // workspace carve-up
  char* ws = (char*)d_ws;
  const size_t f32tok = (size_t)NTOK * DIMV * sizeof(float);     // 50.3 MB
  const size_t bf16tok= (size_t)NTOK * DIMV * sizeof(short);     // 25.2 MB
  float*  x   = (float*)ws;            ws += f32tok;
  float*  qb  = (float*)ws;            ws += f32tok;
  float*  kb  = (float*)ws;            ws += f32tok;
  float*  vb  = (float*)ws;            ws += f32tok;
  float*  rvb = (float*)ws;            ws += f32tok;
  float*  mix = (float*)ws;            ws += (size_t)NTOK * HEADS * sizeof(float);
  float*  gat = (float*)ws;            ws += (size_t)NTOK * HEADS * sizeof(float);
  __bf16* tn  = (__bf16*)ws;           ws += bf16tok;
  __bf16* ob  = (__bf16*)ws;           ws += bf16tok;
  __bf16* act = (__bf16*)ws;           ws += (size_t)NTOK * DFFV * sizeof(short);

  const dim3 blk256(256), blk32(32);
  const dim3 gemm_g768(DIMV / GBN, NTOK / GBM);      // (12,128)
  const dim3 gemm_gff1(DFFV / GBN, NTOK / GBM);      // (32,128)
  const int  th_blocks = NTOK * HEADS;               // 196608

  // x = tokens
  hipMemcpyAsync(x, tokens, f32tok, hipMemcpyDeviceToDevice, stream);

  // value residual: rv = rmsnorm(tokens, vr_norm_w) @ vr_W
  rmsnorm_bf16_kernel<<<NTOK, blk256, 0, stream>>>(tokens, vr_norm_w, tn);
  gemm_kernel<false><<<gemm_g768, blk256, 0, stream>>>(tn, vr_W, nullptr, rvb,
                                                       NTOK, DIMV, DIMV);

  for (int i = 0; i < DEPTH; i++) {
    const int is_time = ((i + 1) % 4 == 0);
    const float* Wq_i = Wq + (size_t)i * DIMV * DIMV;
    const float* Wk_i = Wk + (size_t)i * DIMV * DIMV;
    const float* Wv_i = Wv + (size_t)i * DIMV * DIMV;
    const float* Wo_i = Wo + (size_t)i * DIMV * DIMV;
    const float* Wg_i = Wg + (size_t)i * DIMV * HEADS;
    const float* Wm_i = Wmix + (size_t)i * DIMV * HEADS;
    const float* bm_i = bmix + (size_t)i * HEADS;
    const float* kg_i = kgam + (size_t)i * HEADS * DHV;
    const float* Wi_i = Win + (size_t)i * DIMV * 2 * DFFV;
    const float* bi_i = b_in + (size_t)i * 2 * DFFV;
    const float* Wu_i = Wout + (size_t)i * DFFV * DIMV;
    const float* bo_i = b_out + (size_t)i * DIMV;

    // ---- attention block ----
    rmsnorm_bf16_kernel<<<NTOK, blk256, 0, stream>>>(x, attn_norm_w + (size_t)i * DIMV, tn);
    gemm_kernel<false><<<gemm_g768, blk256, 0, stream>>>(tn, Wq_i, nullptr, qb, NTOK, DIMV, DIMV);
    gemm_kernel<false><<<gemm_g768, blk256, 0, stream>>>(tn, Wk_i, nullptr, kb, NTOK, DIMV, DIMV);
    gemm_kernel<false><<<gemm_g768, blk256, 0, stream>>>(tn, Wv_i, nullptr, vb, NTOK, DIMV, DIMV);
    mixgate_kernel<<<th_blocks, blk32, 0, stream>>>(tn, Wm_i, bm_i, Wg_i, mix, gat);
    vmix_kernel<<<(NTOK * DIMV) / 256, blk256, 0, stream>>>(vb, rvb, mix);
    qkprep_kernel<<<th_blocks, blk32, 0, stream>>>(qb, kb, kg_i, is_time);
    if (is_time) {
      // sequences over t: B*S = 512 sequences, n = 32, stride = S tokens, causal
      size_t shmem = (size_t)TV * DHV * 2 * sizeof(float);   // 16 KB
      attn_kernel<<<BV * SV * HEADS, dim3(TV), shmem, stream>>>(qb, kb, vb, gat, ob,
                                                                TV, SV, 1, 1);
    } else {
      // sequences over s: B*T = 64 sequences, n = 256, stride = 1, non-causal
      size_t shmem = (size_t)SV * DHV * 2 * sizeof(float);   // 128 KB (<320KB WGP LDS)
      attn_kernel<<<BV * TV * HEADS, dim3(SV), shmem, stream>>>(qb, kb, vb, gat, ob,
                                                                SV, 1, 0, 0);
    }
    // x += o @ Wo
    gemm_kernel<true><<<gemm_g768, blk256, 0, stream>>>(ob, Wo_i, nullptr, x, NTOK, DIMV, DIMV);

    // ---- FF block ----
    rmsnorm_bf16_kernel<<<NTOK, blk256, 0, stream>>>(x, ff_norm_w + (size_t)i * DIMV, tn);
    gemm_ff1_kernel<<<gemm_gff1, blk256, 0, stream>>>(tn, Wi_i, bi_i, act);
    gemm_kernel<true><<<gemm_g768, blk256, 0, stream>>>(act, Wu_i, bo_i, x, NTOK, DIMV, DFFV);
  }

  rmsnorm_f32_kernel<<<NTOK, blk256, 0, stream>>>(x, final_norm_w, (float*)d_out);
}
